// HGT_22247930593809
// MI455X (gfx1250) — compile-verified, matrix operand order checked by reading, not run on hardware
//
#include <hip/hip_runtime.h>
#include <hip/hip_bf16.h>
#include <math.h>

typedef __bf16 bf16;
typedef bf16 v16bf __attribute__((ext_vector_type(16)));
typedef bf16 v8bf  __attribute__((ext_vector_type(8)));
typedef float v8f  __attribute__((ext_vector_type(8)));

#define HID   128
#define DH    16
#define NHEAD 8
#define ROWS  64   // rows per block (4 row-tiles of 16)

__device__ __forceinline__ float gelu_exact(float x) {
  return 0.5f * x * (1.f + erff(x * 0.70710678118654752f));
}

__device__ __forceinline__ void atomicMaxFloat(float* addr, float val) {
  if (val >= 0.f) atomicMax((int*)addr, __float_as_int(val));
  else            atomicMin((unsigned int*)addr, __float_as_uint(val));
}

// ---------------- weight packing into wave32 WMMA B-fragment layout --------
// Packed layout: [kt (K/32)][ntile (N/16)][lane (32)][16 bf16]
// lane holds column n = ntile*16 + (lane&15); half j (j<8) -> K = kt*32 + (lane>>4)*8 + j
//                                             half j (j>=8)-> K = kt*32 + 16 + (lane>>4)*8 + (j-8)
__global__ void pack_dense(const float* __restrict__ W, bf16* __restrict__ out, int N) {
  int tid = blockIdx.x * blockDim.x + threadIdx.x;
  if (tid >= 128 * N) return;
  int j    = tid & 15;
  int lane = (tid >> 4) & 31;
  int tile = tid >> 9;
  int NT = N >> 4;
  int kt = tile / NT, nt = tile - kt * NT;
  int n = (nt << 4) + (lane & 15);
  int k = (kt << 5) + ((lane >> 4) << 3) + (j & 7) + ((j & 8) ? 16 : 0);
  out[tid] = (bf16)W[k * N + n];
}

// rel: [8,16,16] -> block-diagonal 128x128 so nhd,hde->nhe becomes a plain GEMM
__global__ void pack_blockdiag(const float* __restrict__ rel, bf16* __restrict__ out) {
  int tid = blockIdx.x * blockDim.x + threadIdx.x;
  if (tid >= 128 * 128) return;
  int j    = tid & 15;
  int lane = (tid >> 4) & 31;
  int tile = tid >> 9;
  int kt = tile >> 3, nt = tile & 7;
  int n = (nt << 4) + (lane & 15);
  int k = (kt << 5) + ((lane >> 4) << 3) + (j & 7) + ((j & 8) ? 16 : 0);
  float v = 0.f;
  if ((k >> 4) == (n >> 4))
    v = rel[(n >> 4) * 256 + (k & 15) * 16 + (n & 15)];
  out[tid] = (bf16)v;
}

// ---------------- fused WMMA GEMM:  out = epi(A[M,128] @ Bpacked[128,N]) ----
// FLAGS bit 0: exact GELU applied to A while staging to LDS
// FLAGS bit 1: ReLU on output
// FLAGS bit 2: sigmoid-gated skip: out = s*y + (1-s)*skip_old, s = sigmoid(*skip_param)
template <int FLAGS>
__global__ __launch_bounds__(256) void gemm_wmma(
    const float* __restrict__ A, const bf16* __restrict__ Bp,
    const float* __restrict__ bias,
    const float* __restrict__ skip_old, const float* __restrict__ skip_param,
    float* __restrict__ out, int M, int N) {
  __shared__ bf16 As[ROWS * HID];               // 16 KB bf16 A-tile (64 rows x K=128)
  const int lane = threadIdx.x & 31;
  const int w    = threadIdx.x >> 5;            // column tile owned by this wave
  const int NT   = N >> 4;
  const int row0 = blockIdx.x << 6;             // 64 rows per block

  // ---- cooperative stage: global f32 -> (gelu) -> bf16 LDS, 8 elems/chunk
  for (int i = threadIdx.x; i < ROWS * 16; i += blockDim.x) {
    int r  = i >> 4;
    int k0 = (i & 15) << 3;
    int gr = row0 + r;
    if (gr >= M) gr = M - 1;
    const float4* src = (const float4*)(A + (size_t)gr * HID + k0);
    float4 f0 = src[0], f1 = src[1];
    float fa[8] = { f0.x, f0.y, f0.z, f0.w, f1.x, f1.y, f1.z, f1.w };
    v8bf pk;
#pragma unroll
    for (int t = 0; t < 8; ++t) {
      float ff = fa[t];
      if (FLAGS & 1) ff = gelu_exact(ff);
      pk[t] = (bf16)ff;
    }
    *(v8bf*)&As[r * HID + k0] = pk;
  }
  __syncthreads();

  const int mloc  = lane & 15;
  const int khalf = (lane >> 4) << 3;           // 0 or 8
  v8f c[4];
#pragma unroll
  for (int rt = 0; rt < 4; ++rt)
    c[rt] = (v8f){0.f, 0.f, 0.f, 0.f, 0.f, 0.f, 0.f, 0.f};

#pragma unroll
  for (int kt = 0; kt < 4; ++kt) {              // K = 128 = 4 x 32
    // one B fragment, reused across 4 row-tiles
    v16bf b = *(const v16bf*)(Bp + ((((size_t)kt * NT + w) << 5) + lane) * 16);
#pragma unroll
    for (int rt = 0; rt < 4; ++rt) {
      const bf16* ap = &As[(rt * 16 + mloc) * HID + (kt << 5) + khalf];
      v8bf lo = *(const v8bf*)ap;
      v8bf hi = *(const v8bf*)(ap + 16);
      v16bf a;
#pragma unroll
      for (int j = 0; j < 8; ++j) { a[j] = lo[j]; a[8 + j] = hi[j]; }
      c[rt] = __builtin_amdgcn_wmma_f32_16x16x32_bf16(false, a, false, b,
                                                      (short)0, c[rt], false, false);
    }
  }

  const int n = (w << 4) + (lane & 15);
  const float bv = bias ? bias[n] : 0.f;
  float gate = 1.f;
  if (FLAGS & 4) {
    float sp = skip_param[0];
    gate = 1.f / (1.f + __expf(-sp));
  }
#pragma unroll
  for (int rt = 0; rt < 4; ++rt) {
    const int rbase = row0 + rt * 16 + ((lane >> 4) << 3);  // C/D: VGPR r -> row rbase+r
    if (rbase + 8 <= M) {                                   // fast path: no guards
#pragma unroll
      for (int r = 0; r < 8; ++r) {
        int m = rbase + r;
        float v = c[rt][r] + bv;
        if (FLAGS & 2) v = fmaxf(v, 0.f);
        if (FLAGS & 4) v = gate * v + (1.f - gate) * skip_old[(size_t)m * N + n];
        out[(size_t)m * N + n] = v;
      }
    } else {
#pragma unroll
      for (int r = 0; r < 8; ++r) {
        int m = rbase + r;
        if (m >= M) continue;
        float v = c[rt][r] + bv;
        if (FLAGS & 2) v = fmaxf(v, 0.f);
        if (FLAGS & 4) v = gate * v + (1.f - gate) * skip_old[(size_t)m * N + n];
        out[(size_t)m * N + n] = v;
      }
    }
  }
}

// ---------------- edge phase: segment softmax + scatter --------------------
__global__ void fill_f32(float* __restrict__ p, float v, size_t n) {
  size_t i = (size_t)blockIdx.x * blockDim.x + threadIdx.x;
  if (i < n) p[i] = v;
}

__global__ void edge_logits_k(const float* __restrict__ q, const float* __restrict__ kt,
                              const int* __restrict__ ei, const float* __restrict__ prel,
                              float* __restrict__ logits, float* __restrict__ mbuf,
                              int E, float scale) {
  int tid = blockIdx.x * blockDim.x + threadIdx.x;
  if (tid >= E * NHEAD) return;
  int e = tid >> 3, h = tid & 7;
  int s = ei[e], d = ei[E + e];
  const float4* qp = (const float4*)(q + (size_t)d * HID + h * DH);
  const float4* kp = (const float4*)(kt + (size_t)s * HID + h * DH);
  float acc = 0.f;
#pragma unroll
  for (int i = 0; i < 4; ++i) {
    float4 a = qp[i], b = kp[i];
    acc += a.x * b.x + a.y * b.y + a.z * b.z + a.w * b.w;
  }
  float lg = acc * prel[h] * scale;
  logits[tid] = lg;
  atomicMaxFloat(mbuf + (size_t)d * NHEAD + h, lg);
}

__global__ void edge_exp_k(const int* __restrict__ ei, float* __restrict__ logits,
                           const float* __restrict__ mbuf, float* __restrict__ den, int E) {
  int tid = blockIdx.x * blockDim.x + threadIdx.x;
  if (tid >= E * NHEAD) return;
  int e = tid >> 3, h = tid & 7;
  int d = ei[E + e];
  float ex = __expf(logits[tid] - mbuf[(size_t)d * NHEAD + h]);
  logits[tid] = ex;
  atomicAdd(den + (size_t)d * NHEAD + h, ex);
}

__global__ void edge_scatter_k(const int* __restrict__ ei, const float* __restrict__ logits,
                               const float* __restrict__ den, const float* __restrict__ vt,
                               float* __restrict__ agg, int E) {
  int tid = blockIdx.x * blockDim.x + threadIdx.x;
  if (tid >= E * NHEAD) return;
  int e = tid >> 3, h = tid & 7;
  int s = ei[e], d = ei[E + e];
  float alpha = logits[tid] / den[(size_t)d * NHEAD + h];
  const float4* vp = (const float4*)(vt + (size_t)s * HID + h * DH);
  float* ag = agg + (size_t)d * HID + h * DH;
#pragma unroll
  for (int i = 0; i < 4; ++i) {
    float4 v = vp[i];
    atomicAdd(ag + i * 4 + 0, v.x * alpha);
    atomicAdd(ag + i * 4 + 1, v.y * alpha);
    atomicAdd(ag + i * 4 + 2, v.z * alpha);
    atomicAdd(ag + i * 4 + 3, v.w * alpha);
  }
}

// ---------------- host orchestration ---------------------------------------
// Param flat order (setup_inputs dict order):
//  0 in_author_W, 1 in_author_b, 2 in_paper_W, 3 in_paper_b
//  per layer l: pb = 4 + 24*l
//    author: pb+0 kW,+1 kb,+2 qW,+3 qb,+4 vW,+5 vb,+6 aW,+7 ab,+8 skip
//    paper : pb+9..+16 (same), +17 skip
//    rels  : +18 arel_ap, +19 mrel_ap, +20 prel_ap, +21 arel_pa, +22 mrel_pa, +23 prel_pa
//  52 out_W, 53 out_b, 54 x_author, 55 x_paper, 56 edge_ap, 57 edge_pa
extern "C" void kernel_launch(void* const* d_in, const int* in_sizes, int n_in,
                              void* d_out, int out_size, void* d_ws, size_t ws_size,
                              hipStream_t stream) {
  (void)n_in; (void)out_size; (void)ws_size;
  const float* xA  = (const float*)d_in[54];
  const float* xP  = (const float*)d_in[55];
  const int*   eAP = (const int*)d_in[56];
  const int*   ePA = (const int*)d_in[57];
  const int NA = in_sizes[54] / HID;
  const int NP = in_sizes[55] / HID;
  const int E  = in_sizes[56] / 2;
  const int Nmax = NA > NP ? NA : NP;

  float* f = (float*)d_ws;
  const size_t NF  = (size_t)Nmax * HID;
  float* Xa[2] = { f + 0 * NF, f + 2 * NF };
  float* Xp[2] = { f + 1 * NF, f + 3 * NF };
  float* Ka = f + 4 * NF;  float* Qa = f + 5 * NF;  float* Va = f + 6 * NF;
  float* Kp = f + 7 * NF;  float* Qp = f + 8 * NF;  float* Vp = f + 9 * NF;
  float* KT = f + 10 * NF; float* VT = f + 11 * NF;
  float* AGa = f + 12 * NF; float* AGp = f + 13 * NF;
  float* LOG = f + 14 * NF;
  float* Mb  = LOG + (size_t)E * NHEAD;
  float* Db  = Mb + (size_t)Nmax * NHEAD;
  bf16*  packB = (bf16*)(Db + (size_t)Nmax * NHEAD);
  auto slot = [&](int i) { return packB + (size_t)i * 16384; };

  // ---- pack all weights to bf16 B-fragment layout (deterministic, each call)
  pack_dense<<<64, 256, 0, stream>>>((const float*)d_in[0], slot(0), HID);
  pack_dense<<<64, 256, 0, stream>>>((const float*)d_in[2], slot(1), HID);
  for (int l = 0; l < 2; ++l) {
    int pb = 4 + 24 * l, sb = 2 + 12 * l;
    int wi[8] = { pb + 0, pb + 2, pb + 4, pb + 6, pb + 9, pb + 11, pb + 13, pb + 15 };
    for (int i = 0; i < 8; ++i)
      pack_dense<<<64, 256, 0, stream>>>((const float*)d_in[wi[i]], slot(sb + i), HID);
    pack_blockdiag<<<64, 256, 0, stream>>>((const float*)d_in[pb + 18], slot(sb + 8));
    pack_blockdiag<<<64, 256, 0, stream>>>((const float*)d_in[pb + 19], slot(sb + 9));
    pack_blockdiag<<<64, 256, 0, stream>>>((const float*)d_in[pb + 21], slot(sb + 10));
    pack_blockdiag<<<64, 256, 0, stream>>>((const float*)d_in[pb + 22], slot(sb + 11));
  }
  pack_dense<<<32, 256, 0, stream>>>((const float*)d_in[52], slot(26), 64);

  auto gemm = [&](const float* A, const bf16* Bp, const float* bias, float* out,
                  const float* so, const float* sp, int M, int N, int flags) {
    dim3 g((M + ROWS - 1) / ROWS), b(2 * N);
    switch (flags) {
      case 0: gemm_wmma<0><<<g, b, 0, stream>>>(A, Bp, bias, so, sp, out, M, N); break;
      case 2: gemm_wmma<2><<<g, b, 0, stream>>>(A, Bp, bias, so, sp, out, M, N); break;
      case 5: gemm_wmma<5><<<g, b, 0, stream>>>(A, Bp, bias, so, sp, out, M, N); break;
      default: break;
    }
  };
  auto edges = [&](const int* ei, const float* qdst, const float* kts, const float* vts,
                   float* agg, const float* prel, int Ndst) {
    size_t nm = (size_t)Ndst * NHEAD;
    fill_f32<<<(int)((nm + 255) / 256), 256, 0, stream>>>(Mb, -INFINITY, nm);
    fill_f32<<<(int)((nm + 255) / 256), 256, 0, stream>>>(Db, 0.f, nm);
    int tE = E * NHEAD;
    edge_logits_k<<<(tE + 255) / 256, 256, 0, stream>>>(qdst, kts, ei, prel, LOG, Mb, E, 0.25f);
    edge_exp_k<<<(tE + 255) / 256, 256, 0, stream>>>(ei, LOG, Mb, Db, E);
    edge_scatter_k<<<(tE + 255) / 256, 256, 0, stream>>>(ei, LOG, Db, vts, agg, E);
  };

  // ---- input projection + ReLU
  gemm(xA, slot(0), (const float*)d_in[1], Xa[0], nullptr, nullptr, NA, HID, 2);
  gemm(xP, slot(1), (const float*)d_in[3], Xp[0], nullptr, nullptr, NP, HID, 2);

  int cur = 0;
  for (int l = 0; l < 2; ++l) {
    int pb = 4 + 24 * l, sb = 2 + 12 * l;
    // k/q/v projections
    gemm(Xa[cur], slot(sb + 0), (const float*)d_in[pb + 1],  Ka, nullptr, nullptr, NA, HID, 0);
    gemm(Xa[cur], slot(sb + 1), (const float*)d_in[pb + 3],  Qa, nullptr, nullptr, NA, HID, 0);
    gemm(Xa[cur], slot(sb + 2), (const float*)d_in[pb + 5],  Va, nullptr, nullptr, NA, HID, 0);
    gemm(Xp[cur], slot(sb + 4), (const float*)d_in[pb + 10], Kp, nullptr, nullptr, NP, HID, 0);
    gemm(Xp[cur], slot(sb + 5), (const float*)d_in[pb + 12], Qp, nullptr, nullptr, NP, HID, 0);
    gemm(Xp[cur], slot(sb + 6), (const float*)d_in[pb + 14], Vp, nullptr, nullptr, NP, HID, 0);

    size_t nfa = (size_t)NA * HID, nfp = (size_t)NP * HID;
    fill_f32<<<(int)((nfa + 255) / 256), 256, 0, stream>>>(AGa, 0.f, nfa);
    fill_f32<<<(int)((nfp + 255) / 256), 256, 0, stream>>>(AGp, 0.f, nfp);

    // author -> paper  (relation transforms via block-diagonal WMMA GEMM)
    gemm(Ka, slot(sb + 8), nullptr, KT, nullptr, nullptr, NA, HID, 0);
    gemm(Va, slot(sb + 9), nullptr, VT, nullptr, nullptr, NA, HID, 0);
    edges(eAP, Qp, KT, VT, AGp, (const float*)d_in[pb + 20], NP);

    // paper -> author
    gemm(Kp, slot(sb + 10), nullptr, KT, nullptr, nullptr, NP, HID, 0);
    gemm(Vp, slot(sb + 11), nullptr, VT, nullptr, nullptr, NP, HID, 0);
    edges(ePA, Qa, KT, VT, AGa, (const float*)d_in[pb + 23], NA);

    // a-projection with GELU on load + sigmoid-gated skip
    gemm(AGa, slot(sb + 3), (const float*)d_in[pb + 7],  Xa[1 - cur],
         Xa[cur], (const float*)d_in[pb + 8],  NA, HID, 5);
    gemm(AGp, slot(sb + 7), (const float*)d_in[pb + 16], Xp[1 - cur],
         Xp[cur], (const float*)d_in[pb + 17], NP, HID, 5);
    cur = 1 - cur;
  }

  // ---- final linear: paper features -> [NP, 64]
  gemm(Xp[cur], slot(26), (const float*)d_in[53], (float*)d_out,
       nullptr, nullptr, NP, 64, 0);
}
